// GNN_DGCNN_60868276519664
// MI455X (gfx1250) — matrix operand order repaired and checked by reading, not compiled, not run on hardware
//
#include <hip/hip_runtime.h>
#include <cstdint>
#include <cstddef>

// ---------------------------------------------------------------------------
// Types for CDNA5 WMMA (wave32): v_wmma_f32_16x16x32_f16
// ---------------------------------------------------------------------------
typedef __attribute__((ext_vector_type(16))) _Float16 v16h;
typedef __attribute__((ext_vector_type(8)))  _Float16 v8h;
typedef __attribute__((ext_vector_type(8)))  float    v8f;

static __device__ inline v8f wmma_f16(v16h a, v16h b, v8f c) {
  // (neg_a, A, neg_b, B, c_mod, C, reuse_a, reuse_b)
  return __builtin_amdgcn_wmma_f32_16x16x32_f16(false, a, false, b, (short)0, c,
                                                false, false);
}
static __device__ inline v8f zero8() {
  v8f z;
#pragma unroll
  for (int i = 0; i < 8; ++i) z[i] = 0.0f;
  return z;
}
static __device__ inline v16h cat8(v8h lo, v8h hi) {
  return __builtin_shufflevector(lo, hi, 0, 1, 2, 3, 4, 5, 6, 7, 8, 9, 10, 11,
                                 12, 13, 14, 15);
}

#define KNN 32
#define HDIM 128

// ---------------------------------------------------------------------------
// Kernel 1: fp32 -> f16 conversion + squared row norms (one wave per row)
// ---------------------------------------------------------------------------
template <int D>
__global__ __launch_bounds__(128) void cvt_norms_kernel(
    const float* __restrict__ f, _Float16* __restrict__ xh,
    float* __restrict__ sq, int n) {
  const int row  = (blockIdx.x * blockDim.x + threadIdx.x) >> 5;
  const int lane = threadIdx.x & 31;
  if (row >= n) return;
  const float* src = f + (size_t)row * D;
  float s = 0.0f;
#pragma unroll
  for (int i = lane; i < D; i += 32) {
    float v = src[i];
    s += v * v;
    xh[(size_t)row * D + i] = (_Float16)v;
  }
#pragma unroll
  for (int off = 16; off; off >>= 1) s += __shfl_xor(s, off);
  if (lane == 0) sq[row] = s;
}

// ---------------------------------------------------------------------------
// Kernel 2: fused kNN. One wave handles 16 query rows vs all n columns.
// Distances via WMMA f16 Gram tiles; running top-32 per row kept in LDS.
// ---------------------------------------------------------------------------
template <int D>
__global__ __launch_bounds__(32) void knn_kernel(
    const _Float16* __restrict__ xh, const float* __restrict__ sq,
    int* __restrict__ knn_idx, int n) {
  __shared__ float ldist[16 * 256];
  __shared__ float bd[16 * 32];
  __shared__ int   bi[16 * 32];

  const int lane = threadIdx.x;
  const int half = lane >> 4;
  const int r    = lane & 15;
  const int rowBase = blockIdx.x * 16;
  constexpr int KS = D / 32;

  for (int t = lane; t < 16 * 32; t += 32) { bd[t] = 3.0e38f; bi[t] = 0; }

  // A fragments (16 query rows), loaded once, reused for every chunk.
  // A layout (16-bit 16x32): lane<16 holds row r, K {0..7,16..23};
  // lane>=16 holds row r, K {8..15,24..31}.
  v16h afrag[KS];
  {
    const _Float16* xrow = xh + (size_t)(rowBase + r) * D;
#pragma unroll
    for (int t = 0; t < KS; ++t) {
      const _Float16* p = xrow + t * 32 + half * 8;
      v8h lo = *(const v8h*)(p);
      v8h hi = *(const v8h*)(p + 16);
      afrag[t] = cat8(lo, hi);
    }
  }
  float sqrow[8];
#pragma unroll
  for (int v = 0; v < 8; ++v) sqrow[v] = sq[rowBase + half * 8 + v];

  for (int cstart = 0; cstart < n; cstart += 256) {
    if (cstart + 256 < n)  // prefetch next chunk's feature rows
      __builtin_prefetch(xh + (size_t)(cstart + 256 + lane) * D, 0, 0);

#pragma unroll 1
    for (int ct = 0; ct < 16; ++ct) {
      const int cb = cstart + ct * 16;
      v8f acc = zero8();
#pragma unroll
      for (int t = 0; t < KS; ++t) {
        // B layout (16-bit 32x16): lane<16 = col r K {0..7,16..23}, lane>=16 = col r K {8..15,24..31}
        const _Float16* p = xh + (size_t)(cb + r) * D + t * 32 + half * 8;
        v16h bfrag = cat8(*(const v8h*)p, *(const v8h*)(p + 16));
        acc = wmma_f16(afrag[t], bfrag, acc);
      }
      const float sqc = sq[cb + r];
#pragma unroll
      for (int v = 0; v < 8; ++v) {
        // D layout: lane holds col r, rows half*8+v
        ldist[(half * 8 + v) * 256 + ct * 16 + r] =
            sqrow[v] + sqc - 2.0f * acc[v];
      }
    }
    __syncthreads();
    if (lane < 16) {
      float* myd = &bd[lane * 32];
      int*   myi = &bi[lane * 32];
      const float* drow = &ldist[lane * 256];
      float cm = myd[0];
      for (int t = 1; t < 32; ++t) cm = fmaxf(cm, myd[t]);
      for (int c = 0; c < 256; ++c) {
        float dv = drow[c];
        if (dv < cm) {  // replace current worst
          int p = 0;
          float mv = myd[0];
          for (int t = 1; t < 32; ++t) {
            float q = myd[t];
            if (q > mv) { mv = q; p = t; }
          }
          myd[p] = dv;
          myi[p] = cstart + c;
          cm = myd[0];
          for (int t = 1; t < 32; ++t) cm = fmaxf(cm, myd[t]);
        }
      }
    }
    __syncthreads();
  }
  if (lane < 16) {
    for (int t = 0; t < 32; ++t)
      knn_idx[(size_t)(rowBase + lane) * KNN + t] = bi[lane * 32 + t];
  }
}

// ---------------------------------------------------------------------------
// Kernel 3: EdgeConv. One wave per node: gather [x_i, x_j - x_i] (32 x 2*DIN),
// GEMM1 (WMMA, w1 in LDS f16) + relu, GEMM2 (w2 in LDS f16) + bias,
// row-wise log_softmax, column max -> 128 outputs.
// ---------------------------------------------------------------------------
template <int DIN>
__global__ __launch_bounds__(128) void edgeconv_kernel(
    const float* __restrict__ feat, const int* __restrict__ knn_idx,
    const float* __restrict__ w1, const float* __restrict__ b1,
    const float* __restrict__ w2, const float* __restrict__ b2,
    float* __restrict__ fout, int n) {
  constexpr int K2 = 2 * DIN;
  constexpr int H  = HDIM;
  extern __shared__ char smem[];
  _Float16* w1t = (_Float16*)smem;        // [H][K2]  (transposed: feature-major)
  _Float16* w2t = w1t + (size_t)H * K2;   // [H][H]
  float* b1s = (float*)(w2t + (size_t)H * H);
  float* b2s = b1s + H;
  float* wavescr = b2s + H;               // per wave: 32*H floats + 32 floats

  const int tid  = threadIdx.x;
  const int wid  = tid >> 5;
  const int lane = tid & 31;
  const int half = lane >> 4;
  const int r    = lane & 15;

  // Cooperative weight staging (transpose so B-fragments are contiguous).
  for (int i = tid; i < K2 * H; i += blockDim.x) {
    int k = i / H, c = i % H;
    w1t[(size_t)c * K2 + k] = (_Float16)w1[i];
  }
  for (int i = tid; i < H * H; i += blockDim.x) {
    int k = i / H, c = i % H;
    w2t[(size_t)c * H + k] = (_Float16)w2[i];
  }
  for (int i = tid; i < H; i += blockDim.x) {
    b1s[i] = b1[i];
    b2s[i] = b2[i];
  }
  __syncthreads();

  const int node = blockIdx.x * 4 + wid;
  if (node >= n) return;

  float*    h2 = wavescr + (size_t)wid * (32 * H + 32);
  _Float16* h1 = (_Float16*)h2;  // overlay (h1 dead before h2 written)
  float* lsebuf = h2 + 32 * H;

  const float* xi = feat + (size_t)node * DIN;
  const int nb0 = knn_idx[(size_t)node * KNN + r];
  const int nb1 = knn_idx[(size_t)node * KNN + 16 + r];
  const float* xj0 = feat + (size_t)nb0 * DIN;
  const float* xj1 = feat + (size_t)nb1 * DIN;

  constexpr int KS1 = K2 / 32;
  v8f acc1[2][8];
#pragma unroll
  for (int m = 0; m < 2; ++m)
#pragma unroll
    for (int nt = 0; nt < 8; ++nt) acc1[m][nt] = zero8();

#pragma unroll
  for (int t = 0; t < KS1; ++t) {
    const bool diff = (t * 32) >= DIN;  // concat halves align with K-steps
    const int kb = diff ? t * 32 - DIN : t * 32;
    v16h a[2];
#pragma unroll
    for (int m = 0; m < 2; ++m) {
      const float* xj = m ? xj1 : xj0;
      v16h av;
#pragma unroll
      for (int e = 0; e < 8; ++e) {
        int k0 = kb + half * 8 + e;
        int k1 = k0 + 16;
        float v0 = diff ? (xj[k0] - xi[k0]) : xi[k0];
        float v1 = diff ? (xj[k1] - xi[k1]) : xi[k1];
        av[e]     = (_Float16)v0;
        av[e + 8] = (_Float16)v1;
      }
      a[m] = av;
    }
#pragma unroll
    for (int nt = 0; nt < 8; ++nt) {
      const _Float16* bp = w1t + (size_t)(nt * 16 + r) * K2 + t * 32 + half * 8;
      v16h bfrag = cat8(*(const v8h*)bp, *(const v8h*)(bp + 16));
      acc1[0][nt] = wmma_f16(a[0], bfrag, acc1[0][nt]);
      acc1[1][nt] = wmma_f16(a[1], bfrag, acc1[1][nt]);
    }
  }

  // bias + relu -> h1 (f16, row-major [32][128]) for GEMM2 A-fragments
#pragma unroll
  for (int nt = 0; nt < 8; ++nt) {
    const int col = nt * 16 + r;
    const float bb = b1s[col];
#pragma unroll
    for (int m = 0; m < 2; ++m)
#pragma unroll
      for (int v = 0; v < 8; ++v) {
        float hv = acc1[m][nt][v] + bb;
        hv = hv > 0.0f ? hv : 0.0f;
        h1[(size_t)(m * 16 + half * 8 + v) * H + col] = (_Float16)hv;
      }
  }

  v8f acc2[2][8];
#pragma unroll
  for (int m = 0; m < 2; ++m)
#pragma unroll
    for (int nt = 0; nt < 8; ++nt) acc2[m][nt] = zero8();

#pragma unroll
  for (int t = 0; t < H / 32; ++t) {
    v16h a[2];
#pragma unroll
    for (int m = 0; m < 2; ++m) {
      const _Float16* hrow = h1 + (size_t)(m * 16 + r) * H + t * 32 + half * 8;
      v8h lo = *(const v8h*)hrow;
      v8h hi = *(const v8h*)(hrow + 16);
      a[m] = cat8(lo, hi);
    }
#pragma unroll
    for (int nt = 0; nt < 8; ++nt) {
      const _Float16* bp = w2t + (size_t)(nt * 16 + r) * H + t * 32 + half * 8;
      v16h bfrag = cat8(*(const v8h*)bp, *(const v8h*)(bp + 16));
      acc2[0][nt] = wmma_f16(a[0], bfrag, acc2[0][nt]);
      acc2[1][nt] = wmma_f16(a[1], bfrag, acc2[1][nt]);
    }
  }

  // bias -> h2 (fp32 [32][128])
#pragma unroll
  for (int nt = 0; nt < 8; ++nt) {
    const int col = nt * 16 + r;
    const float bb = b2s[col];
#pragma unroll
    for (int m = 0; m < 2; ++m)
#pragma unroll
      for (int v = 0; v < 8; ++v)
        h2[(size_t)(m * 16 + half * 8 + v) * H + col] = acc2[m][nt][v] + bb;
  }

  // log_softmax per row (lane owns one of the 32 neighbor rows)
  {
    const float* hr = h2 + (size_t)lane * H;
    float rmax = -3.0e38f;
    for (int f = 0; f < H; ++f) rmax = fmaxf(rmax, hr[f]);
    float s = 0.0f;
    for (int f = 0; f < H; ++f) s += __expf(hr[f] - rmax);
    lsebuf[lane] = rmax + __logf(s);
  }

  // column-wise max over 32 rows; lane handles 4 features
#pragma unroll
  for (int q = 0; q < 4; ++q) {
    const int f = lane * 4 + q;
    float mx = -3.0e38f;
    for (int row = 0; row < 32; ++row)
      mx = fmaxf(mx, h2[(size_t)row * H + f] - lsebuf[row]);
    fout[(size_t)node * H + f] = mx;
  }
}

// ---------------------------------------------------------------------------
// Kernel 4: final linear 128->10 + softmax (one wave per node)
// ---------------------------------------------------------------------------
__global__ __launch_bounds__(128) void classify_kernel(
    const float* __restrict__ feat, const float* __restrict__ w,
    const float* __restrict__ b, float* __restrict__ out, int n) {
  const int node = (blockIdx.x * blockDim.x + threadIdx.x) >> 5;
  const int lane = threadIdx.x & 31;
  if (node >= n) return;
  const float* fr = feat + (size_t)node * HDIM;
  float acc[10];
#pragma unroll
  for (int c = 0; c < 10; ++c) acc[c] = 0.0f;
  for (int f = lane; f < HDIM; f += 32) {
    float x = fr[f];
#pragma unroll
    for (int c = 0; c < 10; ++c) acc[c] += x * w[f * 10 + c];
  }
#pragma unroll
  for (int c = 0; c < 10; ++c)
#pragma unroll
    for (int off = 16; off; off >>= 1) acc[c] += __shfl_xor(acc[c], off);
  if (lane == 0) {
    float mx = -3.0e38f;
#pragma unroll
    for (int c = 0; c < 10; ++c) {
      acc[c] += b[c];
      mx = fmaxf(mx, acc[c]);
    }
    float s = 0.0f;
#pragma unroll
    for (int c = 0; c < 10; ++c) {
      acc[c] = __expf(acc[c] - mx);
      s += acc[c];
    }
    const float inv = 1.0f / s;
#pragma unroll
    for (int c = 0; c < 10; ++c) out[(size_t)node * 10 + c] = acc[c] * inv;
  }
}

// ---------------------------------------------------------------------------
// Host-side launch
// ---------------------------------------------------------------------------
static size_t ec_shm_bytes(int din) {
  size_t K2 = 2 * (size_t)din;
  return (128 * K2 + 128 * 128) * 2 + 256 * 4 + 4 * ((size_t)32 * 128 + 32) * 4;
}

extern "C" void kernel_launch(void* const* d_in, const int* in_sizes, int n_in,
                              void* d_out, int out_size, void* d_ws,
                              size_t ws_size, hipStream_t stream) {
  (void)n_in; (void)out_size; (void)ws_size;
  const float* x      = (const float*)d_in[0];
  // d_in[1] edge_index: unused (graph is rebuilt dynamically)
  const float* l0_w1 = (const float*)d_in[2];
  const float* l0_b1 = (const float*)d_in[3];
  const float* l0_w2 = (const float*)d_in[4];
  const float* l0_b2 = (const float*)d_in[5];
  const float* l1_w1 = (const float*)d_in[6];
  const float* l1_b1 = (const float*)d_in[7];
  const float* l1_w2 = (const float*)d_in[8];
  const float* l1_b2 = (const float*)d_in[9];
  const float* l2_w1 = (const float*)d_in[10];
  const float* l2_b1 = (const float*)d_in[11];
  const float* l2_w2 = (const float*)d_in[12];
  const float* l2_b2 = (const float*)d_in[13];
  const float* out_w = (const float*)d_in[14];
  const float* out_b = (const float*)d_in[15];

  const int n = in_sizes[0] / 64;  // x is [n, 64]

  // Workspace layout
  char* p = (char*)d_ws;
  _Float16* xh = (_Float16*)p;  p += (size_t)n * 128 * sizeof(_Float16);
  float* sqn   = (float*)p;     p += (size_t)n * sizeof(float);
  int*   knn   = (int*)p;       p += (size_t)n * KNN * sizeof(int);
  float* fA    = (float*)p;     p += (size_t)n * HDIM * sizeof(float);
  float* fB    = (float*)p;

  const size_t shm64  = ec_shm_bytes(64);
  const size_t shm128 = ec_shm_bytes(128);
  hipFuncSetAttribute(reinterpret_cast<const void*>(edgeconv_kernel<64>),
                      hipFuncAttributeMaxDynamicSharedMemorySize, (int)shm64);
  hipFuncSetAttribute(reinterpret_cast<const void*>(edgeconv_kernel<128>),
                      hipFuncAttributeMaxDynamicSharedMemorySize, (int)shm128);

  const int wavesBlk = (n + 3) / 4;   // 4 waves (=4 rows/nodes) per 128-thr block
  const int knnBlk   = (n + 15) / 16;

  // ---- Layer 0 (d_in = 64) ----
  cvt_norms_kernel<64><<<wavesBlk, 128, 0, stream>>>(x, xh, sqn, n);
  knn_kernel<64><<<knnBlk, 32, 0, stream>>>(xh, sqn, knn, n);
  edgeconv_kernel<64><<<wavesBlk, 128, shm64, stream>>>(
      x, knn, l0_w1, l0_b1, l0_w2, l0_b2, fA, n);

  // ---- Layer 1 (d_in = 128) ----
  cvt_norms_kernel<128><<<wavesBlk, 128, 0, stream>>>(fA, xh, sqn, n);
  knn_kernel<128><<<knnBlk, 32, 0, stream>>>(xh, sqn, knn, n);
  edgeconv_kernel<128><<<wavesBlk, 128, shm128, stream>>>(
      fA, knn, l1_w1, l1_b1, l1_w2, l1_b2, fB, n);

  // ---- Layer 2 (d_in = 128) ----
  cvt_norms_kernel<128><<<wavesBlk, 128, 0, stream>>>(fB, xh, sqn, n);
  knn_kernel<128><<<knnBlk, 32, 0, stream>>>(xh, sqn, knn, n);
  edgeconv_kernel<128><<<wavesBlk, 128, shm128, stream>>>(
      fB, knn, l2_w1, l2_b1, l2_w2, l2_b2, fA, n);

  // ---- Output head ----
  classify_kernel<<<wavesBlk, 128, 0, stream>>>(fA, out_w, out_b,
                                                (float*)d_out, n);
}
